// TSGRU_81681688035906
// MI455X (gfx1250) — matrix operand order repaired and checked by reading, not compiled, not run on hardware
//
#include <hip/hip_runtime.h>

typedef __attribute__((ext_vector_type(16))) __bf16 v16bf;
typedef __attribute__((ext_vector_type(8)))  float  v8f;
typedef __attribute__((ext_vector_type(4)))  unsigned int v4u;

#define BATCH 64
#define SEQ   512
#define DIM   1024
#define HID   1024
#define LYR   4
#define G3    3072

#define NBLK  96
#define NTHR  256

// packed activation: [row_tile(4)][k_tile(32)][lane(32)][half(16)] bf16 = 65536 elems
#define APACK (4 * 32 * 32 * 16)

// ---- A-matrix (16-bit, 16x32) lane/half position per CDNA5 ISA 7.12.2 ----
__device__ __forceinline__ int apos(int rt, int M, int k) {
  int kt = k >> 5, kk = k & 31;
  int ln = M + (((kk >> 3) & 1) ? 16 : 0);           // lanes 0-15: K 0-7,16-23 ; 16-31: K 8-15,24-31
  int hf = (kk & 7) + ((kk & 16) ? 8 : 0);
  return ((rt * 32 + kt) * 32 + ln) * 16 + hf;
}

__device__ __forceinline__ float sigm(float v)  { return 1.0f / (1.0f + __expf(-v)); }
__device__ __forceinline__ float tanhf_(float v){ float e = __expf(2.0f * v); return 1.0f - 2.0f / (e + 1.0f); }

__device__ __forceinline__ void ld8(const float* base, int tile, int lane, float v[8]) {
  const float4* p = (const float4*)(base + (size_t)tile * 256 + lane * 8);
  float4 a = p[0], b = p[1];
  v[0]=a.x; v[1]=a.y; v[2]=a.z; v[3]=a.w; v[4]=b.x; v[5]=b.y; v[6]=b.z; v[7]=b.w;
}

// device-wide barrier: monotonic counter, s_sleep spin, agent-scope fences
__device__ __forceinline__ void gbar(unsigned* bar, unsigned* phase) {
  __builtin_amdgcn_fence(__ATOMIC_RELEASE, "agent");
  __syncthreads();
  ++(*phase);
  if (threadIdx.x == 0) {
    __hip_atomic_fetch_add(bar, 1u, __ATOMIC_RELAXED, __HIP_MEMORY_SCOPE_AGENT);
    const unsigned tgt = (*phase) * NBLK;
    while (__hip_atomic_load(bar, __ATOMIC_RELAXED, __HIP_MEMORY_SCOPE_AGENT) < tgt)
      __builtin_amdgcn_s_sleep(2);
  }
  __syncthreads();
  __builtin_amdgcn_fence(__ATOMIC_ACQUIRE, "agent");
}

// ---------------- pre-pass kernels ----------------
__global__ void tsgru_init(const float* __restrict__ tau, float* __restrict__ inv_tau,
                           float* __restrict__ hdf, __bf16* __restrict__ hdp,
                           unsigned* __restrict__ bar) {
  const int stride = gridDim.x * blockDim.x;
  int i = blockIdx.x * blockDim.x + threadIdx.x;
  if (i == 0) *bar = 0u;
  if (i < LYR * HID) inv_tau[i] = 1.0f / logf(1.0f + __expf(tau[i]));   // 1/softplus
  for (int j = i; j < LYR * BATCH * HID; j += stride) hdf[j] = 0.0f;    // h0 = 0 -> hd = 0
  for (int j = i; j < LYR * APACK; j += stride) hdp[j] = (__bf16)0.0f;
}

// fp32 weights -> bf16 packed WMMA B-layout: [layer][mat][kt(32)][nt(192)][lane(32)][half(16)]
__global__ void tsgru_pack_w(const float* __restrict__ W_ih, const float* __restrict__ W_hh,
                             __bf16* __restrict__ wp) {
  long long idx = (long long)blockIdx.x * blockDim.x + threadIdx.x;
  if (idx >= (long long)LYR * 2 * G3 * DIM) return;
  int k = (int)(idx & (DIM - 1));
  int j = (int)((idx >> 10) % G3);
  int rest = (int)(idx / ((long long)DIM * G3));
  int mat = rest & 1, layer = rest >> 1;
  const float* src = mat ? W_hh : W_ih;
  float v = src[(size_t)layer * G3 * DIM + (size_t)j * DIM + k];
  int kt = k >> 5, kk = k & 31, nt = j >> 4, nn = j & 15;
  int ln = nn + (((kk >> 3) & 1) ? 16 : 0);         // B 32x16: lanes 0-15 K 0-7,16-23 ; 16-31 K 8-15,24-31
  int hf = (kk & 7) + ((kk & 16) ? 8 : 0);
  size_t dst = ((((size_t)(layer * 2 + mat) * 32 + kt) * 192 + nt) * 32 + ln) * 16 + hf;
  wp[dst] = (__bf16)v;
}

__global__ void tsgru_pack_x(const float* __restrict__ x, __bf16* __restrict__ inp, int t) {
  int e = blockIdx.x * blockDim.x + threadIdx.x;    // 65536 elems
  int row = e >> 10, k = e & 1023;
  float v = x[(size_t)row * SEQ * DIM + (size_t)t * DIM + k];
  inp[apos(row >> 4, row & 15, k)] = (__bf16)v;
}

// ---------------- persistent recurrent kernel ----------------
__global__ void __launch_bounds__(NTHR)
tsgru_persist(const float* __restrict__ x, const float* __restrict__ dt,
              const float* __restrict__ b_ih, const float* __restrict__ b_hh,
              const __bf16* __restrict__ wpack, const float* __restrict__ inv_tau,
              float* __restrict__ hd_f32, __bf16* __restrict__ hd_pack,
              __bf16* __restrict__ inp_pack,
              float* __restrict__ gx, float* __restrict__ gh,
              float* __restrict__ out, unsigned* __restrict__ bar) {
  __shared__ v4u smem[2][2][128];                   // [buf][inp/hd][2 k-slices x 64 chunks] = 8KB
  const int tid  = threadIdx.x;
  const int wv   = tid >> 5;
  const int lane = tid & 31;
  const int rtB  = blockIdx.x / 24;                 // batch row tile 0..3 (8 waves share A)
  const int ctB  = (blockIdx.x % 24) * 8 + wv;      // gate col tile 0..191
  const int gw   = blockIdx.x * 8 + wv;             // global wave id 0..767
  const int mi   = tid >> 7;                        // async loader: matrix 0=inp, 1=hd
  const int c7   = tid & 127;                       // async loader: 16B chunk of 2KB stage
  unsigned phase = 0;

  #pragma unroll 1
  for (int t = 0; t < SEQ; ++t) {
    #pragma unroll 1
    for (int layer = 0; layer < LYR; ++layer) {
      const __bf16* hdP = hd_pack + (size_t)layer * APACK;

      // all 256 threads issue exactly one async b128: four 16x32 A-tiles (2 inp + 2 hd) -> LDS
      auto issue = [&](int p, int buf) {
        const __bf16* src = (mi ? hdP : inp_pack) +
                            (size_t)(rtB * 32 + 2 * p) * 512 + c7 * 8;
        unsigned ldst = (unsigned)(unsigned long long)(void*)&smem[buf][mi][c7];
        unsigned long long ga = (unsigned long long)src;
        asm volatile("global_load_async_to_lds_b128 %0, %1, off"
                     :: "v"(ldst), "v"(ga) : "memory");
      };

      // ---------------- Stage B: dual GEMM, one 16x16 tile per wave ----------------
      v8f ax = {0,0,0,0,0,0,0,0};
      v8f ah = {0,0,0,0,0,0,0,0};
      // running weight pointers; kt stride = 192*512 bf16 = 12288 v4u
      const v4u* wI = (const v4u*)wpack + ((size_t)(layer * 2 + 0) * 32 * 192 + ctB) * 64 + lane * 2;
      const v4u* wH = (const v4u*)wpack + ((size_t)(layer * 2 + 1) * 32 * 192 + ctB) * 64 + lane * 2;
      issue(0, 0);
      #pragma unroll 1
      for (int p = 0; p < 16; ++p) {
        int buf = p & 1;
        if (p < 15) {
          issue(p + 1, buf ^ 1);
          asm volatile("s_wait_asynccnt 0x1" ::: "memory");
        } else {
          asm volatile("s_wait_asynccnt 0x0" ::: "memory");
        }
        __syncthreads();
        union { v4u q[2]; v16bf v; } a0, a1, h0, h1, bI0, bI1, bH0, bH1;
        a0.q[0] = smem[buf][0][lane * 2];      a0.q[1] = smem[buf][0][lane * 2 + 1];
        h0.q[0] = smem[buf][1][lane * 2];      h0.q[1] = smem[buf][1][lane * 2 + 1];
        a1.q[0] = smem[buf][0][64 + lane * 2]; a1.q[1] = smem[buf][0][64 + lane * 2 + 1];
        h1.q[0] = smem[buf][1][64 + lane * 2]; h1.q[1] = smem[buf][1][64 + lane * 2 + 1];
        // B tiles stream once per layer-step out of L2: non-temporal
        bI0.q[0] = __builtin_nontemporal_load(wI + 0);
        bI0.q[1] = __builtin_nontemporal_load(wI + 1);
        bH0.q[0] = __builtin_nontemporal_load(wH + 0);
        bH0.q[1] = __builtin_nontemporal_load(wH + 1);
        bI1.q[0] = __builtin_nontemporal_load(wI + 12288);
        bI1.q[1] = __builtin_nontemporal_load(wI + 12289);
        bH1.q[0] = __builtin_nontemporal_load(wH + 12288);
        bH1.q[1] = __builtin_nontemporal_load(wH + 12289);
        ax = __builtin_amdgcn_wmma_f32_16x16x32_bf16(false, a0.v, false, bI0.v, (short)0, ax, false, false);
        ah = __builtin_amdgcn_wmma_f32_16x16x32_bf16(false, h0.v, false, bH0.v, (short)0, ah, false, false);
        ax = __builtin_amdgcn_wmma_f32_16x16x32_bf16(false, a1.v, false, bI1.v, (short)0, ax, false, false);
        ah = __builtin_amdgcn_wmma_f32_16x16x32_bf16(false, h1.v, false, bH1.v, (short)0, ah, false, false);
        wI += 24576;  wH += 24576;              // advance two kt slices
        __syncthreads();                        // stage-buffer reuse fence
      }
      {
        int col = ctB * 16 + (lane & 15);
        float bx = b_ih[layer * G3 + col];
        float bh = b_hh[layer * G3 + col];
        float* gxp = gx + (size_t)(rtB * 192 + ctB) * 256 + lane * 8;
        float* ghp = gh + (size_t)(rtB * 192 + ctB) * 256 + lane * 8;
        *(float4*)(gxp)     = make_float4(ax[0]+bx, ax[1]+bx, ax[2]+bx, ax[3]+bx);
        *(float4*)(gxp + 4) = make_float4(ax[4]+bx, ax[5]+bx, ax[6]+bx, ax[7]+bx);
        *(float4*)(ghp)     = make_float4(ah[0]+bh, ah[1]+bh, ah[2]+bh, ah[3]+bh);
        *(float4*)(ghp + 4) = make_float4(ah[4]+bh, ah[5]+bh, ah[6]+bh, ah[7]+bh);
      }
      gbar(bar, &phase);

      // ---------------- Stage C: gates + repack next inputs ----------------
      if (gw < 256) {
        int rt = gw >> 6, c = gw & 63;
        int col = c * 16 + (lane & 15);
        int rbase = rt * 16 + ((lane >> 4) << 3);
        float xr[8], xz[8], xn[8], hr[8], hz[8], hn[8];
        ld8(gx, rt * 192 +        c, lane, xr);
        ld8(gx, rt * 192 +  64 +  c, lane, xz);
        ld8(gx, rt * 192 + 128 +  c, lane, xn);
        ld8(gh, rt * 192 +        c, lane, hr);
        ld8(gh, rt * 192 +  64 +  c, lane, hz);
        ld8(gh, rt * 192 + 128 +  c, lane, hn);
        float*  hdF  = hd_f32  + (size_t)layer * BATCH * HID;
        __bf16* hdPk = hd_pack + (size_t)layer * APACK;
        float itau = inv_tau[layer * HID + col];
        #pragma unroll
        for (int i = 0; i < 8; ++i) {
          int row = rbase + i;
          float hd = hdF[row * HID + col];
          float r = sigm(xr[i] + hr[i]);
          float z = sigm(xz[i] + hz[i]);
          float n = tanhf_(xn[i] + r * hn[i]);
          float hnew = (1.0f - z) * n + z * hd;
          if (layer == LYR - 1)
            out[(size_t)row * SEQ * HID + (size_t)t * HID + col] = hnew;
          else
            inp_pack[apos(rt, row & 15, col)] = (__bf16)hnew;   // next layer's A
          if (t + 1 < SEQ) {
            float dec = __expf(-dt[row * SEQ + (t + 1)] * itau);
            float hdn = hnew * dec;
            hdF[row * HID + col] = hdn;
            hdPk[apos(rt, row & 15, col)] = (__bf16)hdn;        // next step's hd A
          }
        }
      } else if (layer == LYR - 1 && gw < 512 && t + 1 < SEQ) {
        int base = (gw - 256) * 256 + lane * 8;                 // pack x[t+1] as layer-0 input
        #pragma unroll
        for (int j = 0; j < 8; ++j) {
          int e = base + j;
          int row = e >> 10, k = e & 1023;
          float v = x[(size_t)row * SEQ * DIM + (size_t)(t + 1) * DIM + k];
          inp_pack[apos(row >> 4, row & 15, k)] = (__bf16)v;
        }
      }
      gbar(bar, &phase);
    }
  }
}

extern "C" void kernel_launch(void* const* d_in, const int* in_sizes, int n_in,
                              void* d_out, int out_size, void* d_ws, size_t ws_size,
                              hipStream_t stream) {
  (void)in_sizes; (void)n_in; (void)out_size; (void)ws_size;
  const float* x    = (const float*)d_in[0];
  const float* dt   = (const float*)d_in[1];
  const float* W_ih = (const float*)d_in[2];
  const float* W_hh = (const float*)d_in[3];
  const float* b_ih = (const float*)d_in[4];
  const float* b_hh = (const float*)d_in[5];
  const float* tau  = (const float*)d_in[6];
  float* out = (float*)d_out;

  char* ws = (char*)d_ws;                    // ~51.2 MB total
  unsigned* bar     = (unsigned*)(ws + 0);
  float*    inv_tau = (float*)   (ws + 256);
  float*    hd_f32  = (float*)   (ws + 16640);
  __bf16*   hd_pack = (__bf16*)  (ws + 1065216);
  __bf16*   inp_pack= (__bf16*)  (ws + 1589504);
  float*    gxb     = (float*)   (ws + 1720576);
  float*    ghb     = (float*)   (ws + 2507008);
  __bf16*   wpack   = (__bf16*)  (ws + 3293440);

  tsgru_init  <<<1024, 256, 0, stream>>>(tau, inv_tau, hd_f32, hd_pack, bar);
  tsgru_pack_w<<<(LYR * 2 * G3 * DIM) / 256, 256, 0, stream>>>(W_ih, W_hh, wpack);
  tsgru_pack_x<<<256, 256, 0, stream>>>(x, inp_pack, 0);
  tsgru_persist<<<NBLK, NTHR, 0, stream>>>(x, dt, b_ih, b_hh, wpack, inv_tau,
                                           hd_f32, hd_pack, inp_pack, gxb, ghb, out, bar);
}
